// BatchedDiffPoolLayer_33028298506469
// MI455X (gfx1250) — compile-verified, matrix operand order checked by reading, not run on hardware
//
#include <hip/hip_runtime.h>
#include <hip/hip_bf16.h>

// Problem constants (match the reference)
#define Nn      50000
#define Ee      800000
#define INF_    128
#define OUTF    128
#define ASSIGN_ 256
#define Bg      32
#define NCOLS   384            // OUTF + ASSIGN
#define KDIM    256            // 2*IN
#define MT      16
#define KT      32
#define NT      16
#define NTILES  (NCOLS / NT)   // 24
#define KTILES  (KDIM / KT)    // 8
#define MTILES  (Nn / MT)      // 3125 (exact)

typedef __attribute__((ext_vector_type(16))) __bf16 v16bf;
typedef __attribute__((ext_vector_type(8)))  float  v8f;

// ---------------------------------------------------------------- zero
__global__ void zero_f32(float* __restrict__ p, int n) {
  int i = blockIdx.x * blockDim.x + threadIdx.x;
  int stride = gridDim.x * blockDim.x;
  for (; i < n; i += stride) p[i] = 0.0f;
}

// ------------------------------------------------- pack W into B-fragments
// B fragment (16x16x32 bf16): lane<16 -> N=lane, holds K {0..7} in halves 0..7,
// K {16..23} in halves 8..15; lane>=16 -> N=lane-16, K {8..15} and {24..31}.
__global__ void pack_w(const float* __restrict__ We, const float* __restrict__ Wp,
                       __bf16* __restrict__ pk) {
  int i = blockIdx.x * blockDim.x + threadIdx.x;
  const int total = KDIM * NCOLS;
  for (; i < total; i += gridDim.x * blockDim.x) {
    int k = i / NCOLS, n = i % NCOLS;
    float w = (n < OUTF) ? We[k * OUTF + n] : Wp[k * ASSIGN_ + (n - OUTF)];
    int nt = n / NT, nn = n % NT;
    int kt = k / KT, kk = k % KT;
    int laneHi = (kk >> 3) & 1;
    int lane   = nn + laneHi * 16;
    int half   = (kk & 7) + ((kk >= 16) ? 8 : 0);
    pk[(((nt * KTILES) + kt) * 32 + lane) * 16 + half] = (__bf16)w;
  }
}

// ---------------------------------------------------------------- degree
__global__ void deg_kernel(const int* __restrict__ dst, float* __restrict__ deg) {
  int e = blockIdx.x * blockDim.x + threadIdx.x;
  if (e < Ee) atomicAdd(&deg[dst[e]], 1.0f);
}

// --------------------------------------------------- neighbor feature scatter
__global__ void scatter_kernel(const float* __restrict__ x, const int* __restrict__ src,
                               const int* __restrict__ dst, float* __restrict__ neigh) {
  long long gid = (long long)blockIdx.x * blockDim.x + threadIdx.x;
  const long long total = (long long)Ee * 32;
  if (gid >= total) return;
  int e    = (int)(gid >> 5);
  int part = (int)(gid & 31);
  int s = src[e], d = dst[e];
  const float4 v = *(const float4*)(x + (long long)s * INF_ + part * 4);
  float* np = neigh + (long long)d * INF_ + part * 4;
  atomicAdd(np + 0, v.x); atomicAdd(np + 1, v.y);
  atomicAdd(np + 2, v.z); atomicAdd(np + 3, v.w);
}

// ------------------- fused bf16-WMMA GEMM + bias + double softmax -> embed, r8
__global__ __launch_bounds__(256) void gemm_fused(
    const float* __restrict__ x, const float* __restrict__ neigh,
    const float* __restrict__ deg, const __bf16* __restrict__ wpk,
    const float* __restrict__ b_embed, const float* __restrict__ b_pool,
    const int* __restrict__ batch, float* __restrict__ embed, float* __restrict__ r8) {
  __shared__ __bf16 Atile[MT * KDIM];   // 8 KB
  __shared__ float  Stage[MT * NCOLS];  // 24 KB
  const int tid   = threadIdx.x;
  const int node0 = blockIdx.x * MT;

  // Stage A tile [16 nodes x 256 K] as bf16; neigh scaled by 1/max(deg,1).
  {
    int row  = tid & 15;
    int cb   = (tid >> 4) * 16;
    int node = node0 + row;
    float scale = 1.0f / fmaxf(deg[node], 1.0f);
#pragma unroll
    for (int j = 0; j < 16; j += 4) {
      int c = cb + j;
      float4 v;
      if (c < 128) {
        v = *(const float4*)(x + (long long)node * INF_ + c);
      } else {
        v = *(const float4*)(neigh + (long long)node * INF_ + (c - 128));
        v.x *= scale; v.y *= scale; v.z *= scale; v.w *= scale;
      }
      Atile[row * KDIM + c + 0] = (__bf16)v.x;
      Atile[row * KDIM + c + 1] = (__bf16)v.y;
      Atile[row * KDIM + c + 2] = (__bf16)v.z;
      Atile[row * KDIM + c + 3] = (__bf16)v.w;
    }
  }
  __syncthreads();

  const int wave   = tid >> 5;
  const int lane   = tid & 31;
  const int laneN  = lane & 15;
  const int laneHi = lane >> 4;

#pragma unroll
  for (int j = 0; j < 3; ++j) {
    const int nt = wave + 8 * j;  // 8 waves x 3 = 24 N-tiles
    v8f acc = {};
#pragma unroll
    for (int kt = 0; kt < KTILES; ++kt) {
      v16bf a, b;
      // A fragment per ISA 16-bit A layout
      const __bf16* Ap = &Atile[laneN * KDIM + kt * 32 + laneHi * 8];
#pragma unroll
      for (int i = 0; i < 8; ++i) { a[i] = Ap[i]; a[8 + i] = Ap[16 + i]; }
      // B fragment: one contiguous 32B region per lane (prepacked)
      const __bf16* Bp = wpk + (((nt * KTILES) + kt) * 32 + lane) * 16;
      if (kt + 1 < KTILES)
        __builtin_prefetch(wpk + (((nt * KTILES) + kt + 1) * 32 + lane) * 16, 0, 1);
#pragma unroll
      for (int i = 0; i < 16; ++i) b[i] = Bp[i];
      acc = __builtin_amdgcn_wmma_f32_16x16x32_bf16(
          false, a, false, b, (short)0, acc, false, false);
    }
    // C layout: lane, vgpr v -> M = v + laneHi*8, N = laneN
#pragma unroll
    for (int v = 0; v < 8; ++v)
      Stage[(v + laneHi * 8) * NCOLS + nt * NT + laneN] = acc[v];
  }
  __syncthreads();

  // embed = C[:, :128] + b_embed
  for (int idx = tid; idx < MT * OUTF; idx += 256) {
    int row = idx >> 7, f = idx & 127;
    embed[(long long)(node0 + row) * OUTF + f] = Stage[row * NCOLS + f] + b_embed[f];
  }

  // pooled = softmax(C[:,128:] + b_pool); r = blockwise renormalized masked softmax.
  // Each wave handles 2 rows; lane ln owns columns [ln*8, ln*8+8) -> graph g's
  // block lives entirely in lane g.
#pragma unroll
  for (int rr = 0; rr < 2; ++rr) {
    int row  = wave * 2 + rr;
    int node = node0 + row;
    float vals[8];
    float mx = -3.4e38f;
#pragma unroll
    for (int j = 0; j < 8; ++j) {
      vals[j] = Stage[row * NCOLS + 128 + lane * 8 + j] + b_pool[lane * 8 + j];
      mx = fmaxf(mx, vals[j]);
    }
    for (int off = 16; off > 0; off >>= 1) mx = fmaxf(mx, __shfl_xor(mx, off));
    float s = 0.0f;
#pragma unroll
    for (int j = 0; j < 8; ++j) { vals[j] = __expf(vals[j] - mx); s += vals[j]; }
    for (int off = 16; off > 0; off >>= 1) s += __shfl_xor(s, off);
    float inv = 1.0f / s;
    int g = batch[node];
    if (lane == g) {
      float w2[8];
      float z2 = 248.0f;  // 256-8 masked columns contribute exp(0)=1 each
#pragma unroll
      for (int j = 0; j < 8; ++j) { w2[j] = __expf(vals[j] * inv); z2 += w2[j]; }
      float sr = 0.0f;
#pragma unroll
      for (int j = 0; j < 8; ++j) { w2[j] /= z2; sr += w2[j]; }
      float rinv = 1.0f / (sr + 1e-13f);
#pragma unroll
      for (int j = 0; j < 8; ++j) r8[(long long)node * 8 + j] = w2[j] * rinv;
    }
  }
}

// -------------------------- h = r^T @ embed via 128KB LDS accumulator
__global__ __launch_bounds__(128) void h_kernel(
    const float* __restrict__ embed, const float* __restrict__ r8,
    const int* __restrict__ batch, float* __restrict__ hout) {
  extern __shared__ float Hacc[];  // ASSIGN_*OUTF floats = 128 KB
  const int tid = threadIdx.x;     // feature index
  for (int i = tid; i < ASSIGN_ * OUTF; i += 128) Hacc[i] = 0.0f;
  __syncthreads();
  const int chunk = (Nn + gridDim.x - 1) / gridDim.x;
  const int i0 = blockIdx.x * chunk;
  const int i1 = min(Nn, i0 + chunk);
  for (int node = i0; node < i1; ++node) {
    int g = batch[node];
    float e = embed[(long long)node * OUTF + tid];
    const float* rp = r8 + (long long)node * 8;
    float* hp = Hacc + (g * 8) * OUTF + tid;  // each thread owns one feature: no races
#pragma unroll
    for (int c = 0; c < 8; ++c) hp[c * OUTF] += rp[c] * e;
  }
  __syncthreads();
  for (int i = tid; i < ASSIGN_ * OUTF; i += 128) atomicAdd(&hout[i], Hacc[i]);
}

// ----------------- adj = r^T A r via per-edge 8x8 outer products, 256KB LDS
__global__ __launch_bounds__(256) void adj_kernel(
    const int* __restrict__ src, const int* __restrict__ dst,
    const float* __restrict__ r8, const int* __restrict__ batch,
    float* __restrict__ adj) {
  extern __shared__ float Aacc[];  // ASSIGN_*ASSIGN_ floats = 256 KB
  const int tid = threadIdx.x;
  for (int i = tid; i < ASSIGN_ * ASSIGN_; i += 256) Aacc[i] = 0.0f;
  __syncthreads();
  const int chunk = (Ee + gridDim.x - 1) / gridDim.x;
  const int e0 = blockIdx.x * chunk;
  const int e1 = min(Ee, e0 + chunk);
  const int sub = tid >> 6;        // 4 edges in flight per 256 threads
  const int a   = (tid >> 3) & 7;
  const int b   = tid & 7;
  for (int eb = e0; eb < e1; eb += 4) {
    int e = eb + sub;
    if (e < e1) {
      int s = src[e], d = dst[e];
      int gs = batch[s], gd = batch[d];
      float val = r8[(long long)s * 8 + a] * r8[(long long)d * 8 + b];
      atomicAdd(&Aacc[(gs * 8 + a) * ASSIGN_ + gd * 8 + b], val);  // ds_add_f32
    }
  }
  __syncthreads();
  for (int i = tid; i < ASSIGN_ * ASSIGN_; i += 256) atomicAdd(&adj[i], Aacc[i]);
}

// ---------------------------------------------------------------- launch
extern "C" void kernel_launch(void* const* d_in, const int* in_sizes, int n_in,
                              void* d_out, int out_size, void* d_ws, size_t ws_size,
                              hipStream_t stream) {
  const float* x     = (const float*)d_in[0];
  const int*   ei    = (const int*)d_in[1];
  const int*   batch = (const int*)d_in[2];
  const float* We    = (const float*)d_in[3];
  const float* be    = (const float*)d_in[4];
  const float* Wp    = (const float*)d_in[5];
  const float* bp    = (const float*)d_in[6];
  const int* src = ei;        // edge_index[0]
  const int* dst = ei + Ee;   // edge_index[1]
  float* out = (float*)d_out; // [adj 256*256][h 256*128]

  // Workspace layout (floats): deg[N] | neigh[N*128] | embed[N*128] | r8[N*8] | Wpack(bf16)
  float* wsf   = (float*)d_ws;
  float* deg   = wsf;
  float* neigh = deg + Nn;
  float* embed = neigh + (size_t)Nn * INF_;
  float* r8    = embed + (size_t)Nn * OUTF;
  __bf16* wpk  = (__bf16*)(r8 + (size_t)Nn * 8);

  // 1. zero deg+neigh (contiguous) and outputs
  zero_f32<<<1024, 256, 0, stream>>>(deg, Nn * (INF_ + 1));
  zero_f32<<<96, 256, 0, stream>>>(out, ASSIGN_ * ASSIGN_ + ASSIGN_ * OUTF);

  // 2. pack weights into WMMA B-fragment order (bf16)
  pack_w<<<96, 256, 0, stream>>>(We, Wp, wpk);

  // 3. degree + neighbor scatter
  deg_kernel<<<(Ee + 255) / 256, 256, 0, stream>>>(dst, deg);
  {
    long long total = (long long)Ee * 32;
    int blocks = (int)((total + 255) / 256);
    scatter_kernel<<<blocks, 256, 0, stream>>>(x, src, dst, neigh);
  }

  // 4. fused GEMM + softmax
  gemm_fused<<<MTILES, 256, 0, stream>>>(x, neigh, deg, wpk, be, bp, batch, embed, r8);

  // 5. h = r^T embed   (128 KB dynamic LDS per block)
  h_kernel<<<64, 128, ASSIGN_ * OUTF * sizeof(float), stream>>>(embed, r8, batch,
                                                                out + ASSIGN_ * ASSIGN_);

  // 6. adj = r^T A r   (256 KB dynamic LDS per block)
  adj_kernel<<<128, 256, ASSIGN_ * ASSIGN_ * sizeof(float), stream>>>(src, dst, r8,
                                                                      batch, out);
}
